// BigramLanguageModel_69166153334827
// MI455X (gfx1250) — compile-verified, matrix operand order checked by reading, not run on hardware
//
#include <hip/hip_runtime.h>

// ---------------- model constants ----------------
#define Bz   64
#define Tz   256
#define Cz   384
#define Hz   6
#define HSz  64
#define Vz   65
#define Lz   6
#define Mz   (Bz * Tz)          // 16384 token rows
#define QKVW 1152               // 3*C, packed q|k|v columns
#define C4   1536               // 4*C

typedef __attribute__((ext_vector_type(16))) _Float16 v16h;
typedef __attribute__((ext_vector_type(8)))  _Float16 v8h;
typedef __attribute__((ext_vector_type(8)))  float    v8f;

__device__ __forceinline__ v8f wmma_f16(v16h a, v16h b, v8f c) {
    return __builtin_amdgcn_wmma_f32_16x16x32_f16(false, a, false, b, (short)0, c, false, false);
}

union Frag16 { v16h v; v8h h2[2]; };
union AccU   { v8f v; float f[8]; };

// ---------------- embedding: x = tok_emb[idx] + pos_emb[t] ----------------
__global__ void embed_kernel(const int* __restrict__ idx,
                             const float* __restrict__ tok_emb,
                             const float* __restrict__ pos_emb,
                             float* __restrict__ x) {
    int i = blockIdx.x * 256 + threadIdx.x;       // over Mz*Cz
    int row = i / Cz, c = i - row * Cz;
    int t = row & (Tz - 1);
    int tok = idx[row];
    x[i] = tok_emb[(size_t)tok * Cz + c] + pos_emb[(size_t)t * Cz + c];
}

// ---------------- layernorm -> f16 (one wave32 per row) ----------------
__global__ void ln_kernel(const float* __restrict__ x,
                          const float* __restrict__ g,
                          const float* __restrict__ b,
                          _Float16* __restrict__ out) {
    int row  = blockIdx.x * 8 + (threadIdx.x >> 5);
    int lane = threadIdx.x & 31;
    const float* xr = x + (size_t)row * Cz;
    float vals[12];
    float s = 0.f, ss = 0.f;
#pragma unroll
    for (int i = 0; i < 12; ++i) {
        float v = xr[lane + i * 32];
        vals[i] = v; s += v; ss += v * v;
    }
#pragma unroll
    for (int off = 16; off; off >>= 1) {
        s  += __shfl_xor(s,  off, 32);
        ss += __shfl_xor(ss, off, 32);
    }
    float mean = s * (1.0f / Cz);
    float var  = ss * (1.0f / Cz) - mean * mean;
    float rstd = rsqrtf(var + 1e-3f);
    _Float16* orow = out + (size_t)row * Cz;
#pragma unroll
    for (int i = 0; i < 12; ++i) {
        int c = lane + i * 32;
        orow[c] = (_Float16)(g[c] * (vals[i] - mean) * rstd + b[c]);
    }
}

// ---------------- weight prep ----------------
// wq/wk/wv are (H, C, HS). Pack as f16, transposed: dst[n][c], n = sec*384 + h*64 + d
__global__ void pack_qkv_kernel(const float* __restrict__ wq,
                                const float* __restrict__ wk,
                                const float* __restrict__ wv,
                                _Float16* __restrict__ dst) {
    int tid = blockIdx.x * 256 + threadIdx.x;     // QKVW*Cz total
    if (tid >= QKVW * Cz) return;
    int n = tid / Cz, c = tid - n * Cz;
    int sec = n / Cz;
    int hn  = n - sec * Cz;
    int h = hn >> 6, d = hn & 63;
    const float* w = (sec == 0) ? wq : ((sec == 1) ? wk : wv);
    dst[(size_t)n * Cz + c] = (_Float16)w[((size_t)h * Cz + c) * HSz + d];
}

// src fp32 [K][N] row-major -> dst f16 [N][K]
__global__ void transposeT_kernel(const float* __restrict__ src,
                                  _Float16* __restrict__ dst, int K, int N) {
    int tid = blockIdx.x * 256 + threadIdx.x;
    if (tid >= K * N) return;
    int n = tid / K, k = tid - n * K;
    dst[tid] = (_Float16)src[(size_t)k * N + n];
}

// ---------------- WMMA GEMM: out[M,N] = A[M,K] * BT[N,K]^T (+bias)(+relu)(+resid) ----
// Block tile 128x128, 256 threads = 8 waves (4 wm x 2 wn); each wave: 32x64 sub-tile
// = 2 A-frags x 4 B-frags -> 8 WMMAs per K=32 step.
// MODE 0: out f16               (QKV, no bias)
// MODE 1: out f16, bias + relu  (MLP-1)
// MODE 2: outf32 += gemm + bias (residual: attn-proj, MLP-2)
template <int MODE>
__global__ void gemm_kernel(const _Float16* __restrict__ A,
                            const _Float16* __restrict__ BT,
                            const float* __restrict__ bias,
                            _Float16* __restrict__ out16,
                            float* __restrict__ outf32,
                            int K, int N) {
    __shared__ __align__(16) _Float16 a_lds[128][32];
    __shared__ __align__(16) _Float16 b_lds[128][32];
    int tid  = threadIdx.x;
    int wave = tid >> 5, lane = tid & 31;
    int wm = wave >> 1, wn = wave & 1;            // 4 x 2 wave grid
    int hlf = lane >> 4, lm = lane & 15;
    int m0 = blockIdx.x * 128, n0 = blockIdx.y * 128;
    int lrow = tid >> 1, lseg = (tid & 1) * 16;   // loader: 2 threads/row, 16 halfs each

    v8f acc[2][4];
#pragma unroll
    for (int i = 0; i < 2; ++i)
#pragma unroll
        for (int t = 0; t < 4; ++t) acc[i][t] = (v8f){};

    for (int kb = 0; kb < K; kb += 32) {
        __syncthreads();
        *(v8h*)&a_lds[lrow][lseg]     = *(const v8h*)(A  + (size_t)(m0 + lrow) * K + kb + lseg);
        *(v8h*)&a_lds[lrow][lseg + 8] = *(const v8h*)(A  + (size_t)(m0 + lrow) * K + kb + lseg + 8);
        *(v8h*)&b_lds[lrow][lseg]     = *(const v8h*)(BT + (size_t)(n0 + lrow) * K + kb + lseg);
        *(v8h*)&b_lds[lrow][lseg + 8] = *(const v8h*)(BT + (size_t)(n0 + lrow) * K + kb + lseg + 8);
        __syncthreads();

        Frag16 af[2], bf[4];
#pragma unroll
        for (int i = 0; i < 2; ++i) {
            int am = wm * 32 + i * 16 + lm;
            af[i].h2[0] = *(const v8h*)&a_lds[am][hlf * 8];
            af[i].h2[1] = *(const v8h*)&a_lds[am][hlf * 8 + 16];
        }
#pragma unroll
        for (int t = 0; t < 4; ++t) {
            int bn = wn * 64 + t * 16 + lm;
            bf[t].h2[0] = *(const v8h*)&b_lds[bn][hlf * 8];
            bf[t].h2[1] = *(const v8h*)&b_lds[bn][hlf * 8 + 16];
        }
#pragma unroll
        for (int i = 0; i < 2; ++i)
#pragma unroll
            for (int t = 0; t < 4; ++t)
                acc[i][t] = wmma_f16(af[i].v, bf[t].v, acc[i][t]);
    }

#pragma unroll
    for (int i = 0; i < 2; ++i)
#pragma unroll
        for (int t = 0; t < 4; ++t) {
            AccU u; u.v = acc[i][t];
            int n = n0 + wn * 64 + t * 16 + lm;
            float bv = (MODE == 0) ? 0.f : bias[n];
#pragma unroll
            for (int r = 0; r < 8; ++r) {
                int m = m0 + wm * 32 + i * 16 + hlf * 8 + r;
                float v = u.f[r] + bv;
                if (MODE == 1) v = v > 0.f ? v : 0.f;
                if (MODE == 2) outf32[(size_t)m * N + n] += v;
                else           out16[(size_t)m * N + n] = (_Float16)v;
            }
        }
}

// ---------------- fused causal attention per (b,h) ----------------
// qkv: f16 [Mz][QKVW] with q|k|v column sections; o: f16 [Mz][Cz] (heads concat)
__global__ void attention_kernel(const _Float16* __restrict__ qkv,
                                 _Float16* __restrict__ o) {
    const int bh = blockIdx.x;
    const int b = bh / Hz, h = bh - b * Hz;
    extern __shared__ __align__(16) char smem[];
    _Float16* k_lds  = (_Float16*)smem;                      // [256][64]  32 KB
    _Float16* vT_lds = (_Float16*)(smem + 32768);            // [64][256]  32 KB
    float*    s_lds  = (float*)(smem + 65536);               // [8][16][256] 128 KB
    _Float16* p_lds  = (_Float16*)(smem + 65536 + 131072);   // [8][16][256] 64 KB

    int tid = threadIdx.x, wave = tid >> 5, lane = tid & 31;
    int hlf = lane >> 4, lm = lane & 15;
    const float scale = 0.05103103630798288f;                // 384^-0.5

    // stage K rows and V transposed
    {
        int t = tid;                                          // 0..255
        const _Float16* krow = qkv + (size_t)(b * Tz + t) * QKVW + Cz     + h * HSz;
        const _Float16* vrow = qkv + (size_t)(b * Tz + t) * QKVW + 2 * Cz + h * HSz;
        _Float16* kd = k_lds + t * 64;
        _Float16 vtmp[64];
#pragma unroll
        for (int i = 0; i < 8; ++i) {
            *(v8h*)(kd + i * 8)   = *(const v8h*)(krow + i * 8);
            *(v8h*)(vtmp + i * 8) = *(const v8h*)(vrow + i * 8);
        }
#pragma unroll
        for (int d = 0; d < 64; ++d) vT_lds[d * 256 + t] = vtmp[d];
    }
    __syncthreads();

    float*    sw = s_lds + wave * 16 * 256;
    _Float16* pw = p_lds + wave * 16 * 256;

    for (int g = 0; g < 2; ++g) {
        int qbase  = g * 128 + wave * 16;
        int qblock = qbase >> 4;
        int ncols  = (qblock + 1) * 16;
        int npad   = (ncols + 31) & ~31;

        Frag16 qf0, qf1;
        const _Float16* qrow = qkv + (size_t)(b * Tz + qbase + lm) * QKVW + h * HSz;
        qf0.h2[0] = *(const v8h*)(qrow + hlf * 8);
        qf0.h2[1] = *(const v8h*)(qrow + hlf * 8 + 16);
        qf1.h2[0] = *(const v8h*)(qrow + 32 + hlf * 8);
        qf1.h2[1] = *(const v8h*)(qrow + 32 + hlf * 8 + 16);

        // S = Q K^T (causal)
        for (int j = 0; j <= qblock; ++j) {
            Frag16 kf0, kf1;
            const _Float16* kr = k_lds + (j * 16 + lm) * 64;
            kf0.h2[0] = *(const v8h*)(kr + hlf * 8);
            kf0.h2[1] = *(const v8h*)(kr + hlf * 8 + 16);
            kf1.h2[0] = *(const v8h*)(kr + 32 + hlf * 8);
            kf1.h2[1] = *(const v8h*)(kr + 32 + hlf * 8 + 16);
            AccU acc; acc.v = (v8f){};
            acc.v = wmma_f16(qf0.v, kf0.v, acc.v);
            acc.v = wmma_f16(qf1.v, kf1.v, acc.v);
            int tk = j * 16 + lm;
#pragma unroll
            for (int r = 0; r < 8; ++r) {
                int rl = hlf * 8 + r;
                float sv = acc.f[r] * scale;
                if (tk > qbase + rl) sv = -3.0e38f;
                sw[rl * 256 + j * 16 + lm] = sv;
            }
        }
        __syncthreads();

        // softmax: lanes l and l^16 split row lm
        {
            int row = lm, c0 = hlf * 128;
            int cend = ncols < c0 + 128 ? ncols : c0 + 128;
            float mx = -3.0e38f;
            for (int c = c0; c < cend; ++c) mx = fmaxf(mx, sw[row * 256 + c]);
            mx = fmaxf(mx, __shfl_xor(mx, 16, 32));
            float sum = 0.f;
            for (int c = c0; c < cend; ++c) sum += __expf(sw[row * 256 + c] - mx);
            sum += __shfl_xor(sum, 16, 32);
            float inv = 1.0f / sum;
            int pend = npad < c0 + 128 ? npad : c0 + 128;
            for (int c = c0; c < pend; ++c) {
                float pv = (c < ncols) ? __expf(sw[row * 256 + c] - mx) * inv : 0.f;
                pw[row * 256 + c] = (_Float16)pv;
            }
        }
        __syncthreads();

        // O = P V
        AccU oacc[4];
#pragma unroll
        for (int dt = 0; dt < 4; ++dt) oacc[dt].v = (v8f){};
        for (int kb = 0; kb < npad; kb += 32) {
            Frag16 pf;
            const _Float16* pr = pw + lm * 256 + kb;
            pf.h2[0] = *(const v8h*)(pr + hlf * 8);
            pf.h2[1] = *(const v8h*)(pr + hlf * 8 + 16);
#pragma unroll
            for (int dt = 0; dt < 4; ++dt) {
                Frag16 vf;
                const _Float16* vr = vT_lds + (dt * 16 + lm) * 256 + kb;
                vf.h2[0] = *(const v8h*)(vr + hlf * 8);
                vf.h2[1] = *(const v8h*)(vr + hlf * 8 + 16);
                oacc[dt].v = wmma_f16(pf.v, vf.v, oacc[dt].v);
            }
        }
#pragma unroll
        for (int dt = 0; dt < 4; ++dt)
#pragma unroll
            for (int r = 0; r < 8; ++r) {
                int t = qbase + hlf * 8 + r;
                int d = dt * 16 + lm;
                o[(size_t)(b * Tz + t) * Cz + h * HSz + d] = (_Float16)oacc[dt].f[r];
            }
        __syncthreads();
    }
}

// ---------------- LM head + per-row loss ----------------
__global__ void lmhead_kernel(const _Float16* __restrict__ hf,
                              const float* __restrict__ wlm,
                              const float* __restrict__ blm,
                              const int* __restrict__ targets,
                              float* __restrict__ logits_out,
                              float* __restrict__ row_loss) {
    int row = blockIdx.x;
    __shared__ float hrow[Cz];
    __shared__ float ls[Vz];
    int tid = threadIdx.x;                         // 128 threads
    for (int i = tid; i < Cz; i += 128) hrow[i] = (float)hf[(size_t)row * Cz + i];
    __syncthreads();
    if (tid < Vz) {
        float acc = blm[tid];
        for (int c = 0; c < Cz; ++c) acc += hrow[c] * wlm[(size_t)c * Vz + tid];
        logits_out[(size_t)row * Vz + tid] = acc;
        ls[tid] = acc;
    }
    __syncthreads();
    if (tid == 0) {
        float mx = -3.0e38f;
        for (int v = 0; v < Vz; ++v) mx = fmaxf(mx, ls[v]);
        float s = 0.f;
        for (int v = 0; v < Vz; ++v) s += __expf(ls[v] - mx);
        float lse = mx + __logf(s);
        row_loss[row] = lse - ls[targets[row]];
    }
}

// deterministic fixed-order loss reduction
__global__ void loss_reduce_kernel(const float* __restrict__ row_loss,
                                   float* __restrict__ out_loss) {
    __shared__ float red[256];
    int tid = threadIdx.x;
    float s = 0.f;
    for (int i = tid; i < Mz; i += 256) s += row_loss[i];
    red[tid] = s; __syncthreads();
    for (int off = 128; off; off >>= 1) {
        if (tid < off) red[tid] += red[tid + off];
        __syncthreads();
    }
    if (tid == 0) *out_loss = red[0] * (1.0f / Mz);
}

// ---------------- host launcher ----------------
extern "C" void kernel_launch(void* const* d_in, const int* in_sizes, int n_in,
                              void* d_out, int out_size, void* d_ws, size_t ws_size,
                              hipStream_t stream) {
    const int*   idx     = (const int*)d_in[0];
    const int*   targets = (const int*)d_in[1];
    const float* tok_emb = (const float*)d_in[2];
    const float* pos_emb = (const float*)d_in[3];
    const float* ln1_g   = (const float*)d_in[4];
    const float* ln1_b   = (const float*)d_in[5];
    const float* wq      = (const float*)d_in[6];
    const float* wk      = (const float*)d_in[7];
    const float* wv      = (const float*)d_in[8];
    const float* wo      = (const float*)d_in[9];
    const float* bo      = (const float*)d_in[10];
    const float* ln2_g   = (const float*)d_in[11];
    const float* ln2_b   = (const float*)d_in[12];
    const float* w1      = (const float*)d_in[13];
    const float* b1      = (const float*)d_in[14];
    const float* w2      = (const float*)d_in[15];
    const float* b2      = (const float*)d_in[16];
    const float* lnf_g   = (const float*)d_in[17];
    const float* lnf_b   = (const float*)d_in[18];
    const float* w_lm    = (const float*)d_in[19];
    const float* b_lm    = (const float*)d_in[20];
    (void)in_sizes; (void)n_in; (void)out_size; (void)ws_size;

    char* ws = (char*)d_ws;
    size_t off = 0;
    auto alloc = [&](size_t bytes) -> void* {
        void* p = ws + off;
        off = (off + bytes + 255) & ~(size_t)255;
        return p;
    };
    float*     x      = (float*)    alloc((size_t)Mz * Cz * 4);
    _Float16*  hbuf   = (_Float16*) alloc((size_t)Mz * Cz * 2);
    _Float16*  qkvbuf = (_Float16*) alloc((size_t)Mz * QKVW * 2);
    _Float16*  obuf   = (_Float16*) alloc((size_t)Mz * Cz * 2);
    _Float16*  mlpbuf = (_Float16*) alloc((size_t)Mz * C4 * 2);
    _Float16*  wqkvT  = (_Float16*) alloc((size_t)QKVW * Cz * 2);
    _Float16*  woT    = (_Float16*) alloc((size_t)Cz * Cz * 2);
    _Float16*  w1T    = (_Float16*) alloc((size_t)C4 * Cz * 2);
    _Float16*  w2T    = (_Float16*) alloc((size_t)Cz * C4 * 2);
    float*     rloss  = (float*)    alloc((size_t)Mz * 4);

    // weight prep (f16, transposed N-major)
    pack_qkv_kernel<<<(QKVW * Cz + 255) / 256, 256, 0, stream>>>(wq, wk, wv, wqkvT);
    transposeT_kernel<<<(Cz * Cz + 255) / 256, 256, 0, stream>>>(wo, woT, Cz, Cz);
    transposeT_kernel<<<(Cz * C4 + 255) / 256, 256, 0, stream>>>(w1, w1T, Cz, C4);
    transposeT_kernel<<<(C4 * Cz + 255) / 256, 256, 0, stream>>>(w2, w2T, C4, Cz);

    embed_kernel<<<(Mz * Cz) / 256, 256, 0, stream>>>(idx, tok_emb, pos_emb, x);

    for (int l = 0; l < Lz; ++l) {
        ln_kernel<<<Mz / 8, 256, 0, stream>>>(x, ln1_g, ln1_b, hbuf);
        gemm_kernel<0><<<dim3(Mz / 128, QKVW / 128), 256, 0, stream>>>(
            hbuf, wqkvT, nullptr, qkvbuf, nullptr, Cz, QKVW);
        attention_kernel<<<Bz * Hz, 256, 262144, stream>>>(qkvbuf, obuf);
        gemm_kernel<2><<<dim3(Mz / 128, Cz / 128), 256, 0, stream>>>(
            obuf, woT, bo, nullptr, x, Cz, Cz);
        ln_kernel<<<Mz / 8, 256, 0, stream>>>(x, ln2_g, ln2_b, hbuf);
        gemm_kernel<1><<<dim3(Mz / 128, C4 / 128), 256, 0, stream>>>(
            hbuf, w1T, b1, mlpbuf, nullptr, Cz, C4);
        gemm_kernel<2><<<dim3(Mz / 128, Cz / 128), 256, 0, stream>>>(
            mlpbuf, w2T, b2, nullptr, x, C4, Cz);
    }

    ln_kernel<<<Mz / 8, 256, 0, stream>>>(x, lnf_g, lnf_b, hbuf);
    lmhead_kernel<<<Mz, 128, 0, stream>>>(hbuf, w_lm, b_lm, targets,
                                          (float*)d_out, rloss);
    loss_reduce_kernel<<<1, 256, 0, stream>>>(rloss, (float*)d_out + (size_t)Mz * Vz);
}
